// CrossAttention_50586124812330
// MI455X (gfx1250) — compile-verified, matrix operand order checked by reading, not run on hardware
//
#include <hip/hip_runtime.h>
#include <hip/hip_bf16.h>

typedef unsigned short u16;
typedef unsigned int   u32;
typedef unsigned long long u64;
typedef __attribute__((ext_vector_type(16))) __bf16 v16bf;
typedef __attribute__((ext_vector_type(8)))  float  v8f;
typedef __attribute__((ext_vector_type(4)))  unsigned v4u;
typedef __attribute__((ext_vector_type(8)))  int      v8i;
typedef __attribute__((ext_vector_type(4)))  int      v4i;

// Feature switches (each degrades to the proven manual path if unavailable).
#if defined(__gfx1250__) && __has_builtin(__builtin_amdgcn_tensor_load_to_lds)
#define TRY_TDM 1
#else
#define TRY_TDM 0
#endif
#if defined(__gfx1250__)
#define TRY_ASYNC 1
#else
#define TRY_ASYNC 0
#endif

__device__ __forceinline__ u32 f2bfbits(float f) {
  u32 u = __builtin_bit_cast(u32, f);
  return (u + 0x7FFFu + ((u >> 16) & 1u)) >> 16;   // round-to-nearest-even
}
__device__ __forceinline__ u32 bfpack2(float lo, float hi) {
  return f2bfbits(lo) | (f2bfbits(hi) << 16);
}
__device__ __forceinline__ v16bf make_frag(const u16* p0, const u16* p1) {
  union { v16bf v; uint4 q[2]; } f;
  f.q[0] = *(const uint4*)p0;
  f.q[1] = *(const uint4*)p1;
  return f.v;
}
__device__ __forceinline__ v8f vzero8() {
  v8f z = {0.f, 0.f, 0.f, 0.f, 0.f, 0.f, 0.f, 0.f};
  return z;
}
__device__ __forceinline__ v8f wmma_bf16(v16bf a, v16bf b, v8f c) {
  return __builtin_amdgcn_wmma_f32_16x16x32_bf16(false, a, false, b, (short)0, c,
                                                 false, false);
}

#if TRY_ASYNC
// Async global -> LDS copy (ASYNCcnt tracked). VDST = per-lane LDS byte addr.
__device__ __forceinline__ void async_load_b128(const void* g, void* lds) {
  u32 l = (u32)(u64)(size_t)lds;                 // addr[31:0] = LDS offset
  u64 ga = (u64)(size_t)g;
  asm volatile("global_load_async_to_lds_b128 %0, %1, off"
               :: "v"(l), "v"(ga) : "memory");
}
__device__ __forceinline__ void wait_async0() {
  asm volatile("s_wait_asynccnt 0x0" ::: "memory");
}
#endif

#if TRY_TDM
// TDM 2D tile load: rows x dim0_elems of 2-byte elements, row stride in elems.
// D# layout per CDNA5 ISA 8.3-8.6 (count=1, type=2 "image", data_size=1 -> 2B).
// This toolchain exposes the 6-arg builtin: (v4u, v8i, v4i, v4i, v8i, cpol).
__device__ __forceinline__ void tdm_load_2d_bf16(void* lds, const void* gaddr,
                                                 u32 dim0_elems, u32 rows,
                                                 u32 stride_elems) {
  u64 ga = (u64)(size_t)gaddr;
  v4u g0;
  g0[0] = 1u;                                     // count=1, is_restore=0
  g0[1] = (u32)(u64)(size_t)lds;                  // lds_addr
  g0[2] = (u32)ga;                                // global_addr[31:0]
  g0[3] = (u32)((ga >> 32) & 0x1FFFFFFu) | (2u << 30);  // addr[56:32] | type=2
  v8i g1;
  g1[0] = (int)(1u << 16);                        // wg_mask=0, data_size=1 (2B)
  g1[1] = (int)(dim0_elems << 16);                // tensor_dim0[15:0]
  g1[2] = (int)(rows << 16);                      // dim0 hi=0 | tensor_dim1[15:0]
  g1[3] = (int)(dim0_elems << 16);                // dim1 hi=0 | tile_dim0
  g1[4] = (int)rows;                              // tile_dim1 | tile_dim2=0
  g1[5] = (int)stride_elems;                      // tensor_dim0_stride[31:0]
  g1[6] = 0;                                      // stride0 hi | stride1 lo
  g1[7] = 0;
  v4i z4 = {0, 0, 0, 0};
  v8i z8 = {0, 0, 0, 0, 0, 0, 0, 0};
  __builtin_amdgcn_tensor_load_to_lds(g0, g1, z4, z4, z8, 0);
}
#endif

// ---------------------------------------------------------------------------
// Kernel 1: qkv = x @ W_qkv  (fp32 -> bf16 out, scattered head-major [B,H,N,d])
// M = 27648 (B*N), K = 768, Ncols = 2304. 128x128 block tile, 8 waves.
// ---------------------------------------------------------------------------
__global__ __launch_bounds__(256) void qkv_gemm_kernel(
    const float* __restrict__ x, const float* __restrict__ W,
    u16* __restrict__ qo, u16* __restrict__ ko, u16* __restrict__ vo)
{
  __shared__ u16 As[128][40];   // [m][k], bf16, padded stride
  __shared__ u16 Bs[128][40];   // [n][k], bf16 (transposed on store)

  const int tid  = threadIdx.x;
  const int lane = tid & 31;
  const int wave = tid >> 5;
  const int half = lane >> 4;
  const int ln   = lane & 15;
  const int wr   = wave >> 1;            // wave row 0..3 (32 rows each)
  const int wc   = wave & 1;             // wave col 0..1 (64 cols each)
  const int rowBase = blockIdx.y * 128;
  const int colBase = blockIdx.x * 128;

  const int ar = tid >> 1;               // 0..127
  const int ac = (tid & 1) << 4;         // 0 or 16
  const int bk = tid >> 3;               // 0..31
  const int bn = (tid & 7) << 4;         // 0..112

  v8f acc[2][4];
#pragma unroll
  for (int mi = 0; mi < 2; ++mi)
#pragma unroll
    for (int ni = 0; ni < 4; ++ni) acc[mi][ni] = vzero8();

  for (int ks = 0; ks < 24; ++ks) {
    const int k0 = ks << 5;
    { // stage A tile (128x32), fp32 -> bf16
      const float4* ap = (const float4*)(x + (size_t)(rowBase + ar) * 768 + k0 + ac);
      float4 f0 = ap[0], f1 = ap[1], f2 = ap[2], f3 = ap[3];
      uint4 p0, p1;
      p0.x = bfpack2(f0.x, f0.y); p0.y = bfpack2(f0.z, f0.w);
      p0.z = bfpack2(f1.x, f1.y); p0.w = bfpack2(f1.z, f1.w);
      p1.x = bfpack2(f2.x, f2.y); p1.y = bfpack2(f2.z, f2.w);
      p1.z = bfpack2(f3.x, f3.y); p1.w = bfpack2(f3.z, f3.w);
      *(uint4*)&As[ar][ac]     = p0;
      *(uint4*)&As[ar][ac + 8] = p1;
    }
    { // stage B tile (32x128) transposed into [n][k]
      const float4* bp = (const float4*)(W + (size_t)(k0 + bk) * 2304 + colBase + bn);
#pragma unroll
      for (int c4 = 0; c4 < 4; ++c4) {
        float4 f = bp[c4];
        Bs[bn + c4 * 4 + 0][bk] = (u16)f2bfbits(f.x);
        Bs[bn + c4 * 4 + 1][bk] = (u16)f2bfbits(f.y);
        Bs[bn + c4 * 4 + 2][bk] = (u16)f2bfbits(f.z);
        Bs[bn + c4 * 4 + 3][bk] = (u16)f2bfbits(f.w);
      }
    }
    if (ks < 23) {
      __builtin_prefetch(x + (size_t)(rowBase + ar) * 768 + k0 + 32 + ac, 0, 3);
      __builtin_prefetch(W + (size_t)(k0 + 32 + bk) * 2304 + colBase + bn, 0, 3);
    }
    __syncthreads();

    v16bf afrag[2], bfrag[4];
#pragma unroll
    for (int mi = 0; mi < 2; ++mi) {
      const u16* r = &As[wr * 32 + mi * 16 + ln][0];
      afrag[mi] = make_frag(r + half * 8, r + 16 + half * 8);
    }
#pragma unroll
    for (int ni = 0; ni < 4; ++ni) {
      const u16* r = &Bs[wc * 64 + ni * 16 + ln][0];
      bfrag[ni] = make_frag(r + half * 8, r + 16 + half * 8);
    }
#pragma unroll
    for (int mi = 0; mi < 2; ++mi)
#pragma unroll
      for (int ni = 0; ni < 4; ++ni)
        acc[mi][ni] = wmma_bf16(afrag[mi], bfrag[ni], acc[mi][ni]);
    __syncthreads();
  }

  // scatter to q/k/v [B,H,N,d] bf16; each 128-col tile lies fully in q, k, or v
  const int which = colBase / 768;
  u16* dst = (which == 0) ? qo : ((which == 1) ? ko : vo);
#pragma unroll
  for (int mi = 0; mi < 2; ++mi)
#pragma unroll
    for (int ni = 0; ni < 4; ++ni)
#pragma unroll
      for (int i = 0; i < 8; ++i) {
        const int rg  = rowBase + wr * 32 + mi * 16 + half * 8 + i;
        const int cg  = colBase + wc * 64 + ni * 16 + ln - which * 768;
        const int b   = rg / 864, tok = rg - b * 864;
        const int hh  = cg >> 6, dd = cg & 63;
        dst[(((size_t)b * 12 + hh) * 864 + tok) * 64 + dd] =
            (u16)f2bfbits(acc[mi][ni][i]);
      }
}

// ---------------------------------------------------------------------------
// Kernels 2/3: cross attention, templated per branch.
// One block = 4 waves = 64 q rows of one (b,h). K staged to LDS by the TDM
// (wave 0 issues tensor_load_to_lds, waits TENSORcnt); V transposed manually.
// ---------------------------------------------------------------------------
template<int KV_LEN, int KV_PAD, int Q_OFF, int Q_LEN, int KV_OFF>
__global__ __launch_bounds__(128) void attn_kernel(
    const u16* __restrict__ q, const u16* __restrict__ k,
    const u16* __restrict__ v, u16* __restrict__ ctx)
{
  constexpr int KV_TILES = KV_PAD / 16;
  constexpr int KSTEPS   = KV_PAD / 32;
  __shared__ u16 Ks[160][64];       // [token][d]
  __shared__ u16 VsT[64][168];      // [d][token] (transposed)
  __shared__ u16 Ps[4][16][168];    // per-wave probabilities

  const int tid  = threadIdx.x;
  const int lane = tid & 31;
  const int wave = tid >> 5;
  const int half = lane >> 4;
  const int ln   = lane & 15;
  const int bh   = blockIdx.y;                 // b*12 + h
  const size_t base = (size_t)bh * 864 * 64;

#if TRY_TDM
  if (wave == 0) {
    tdm_load_2d_bf16(&Ks[0][0], k + base + (size_t)KV_OFF * 64,
                     /*dim0=*/64, /*rows=*/KV_PAD, /*stride=*/64);
    __builtin_amdgcn_s_wait_tensorcnt(0);
  }
#else
  for (int idx = tid; idx < KV_PAD * 8; idx += 128) {         // k: uint4 copies
    const int row = idx >> 3, c8 = (idx & 7) << 3;
    *(uint4*)&Ks[row][c8] =
        *(const uint4*)(k + base + (size_t)(KV_OFF + row) * 64 + c8);
  }
#endif
  for (int idx = tid; idx < KV_PAD * 64; idx += 128) {        // v: transpose
    const int row = idx >> 6, col = idx & 63;
    VsT[col][row] = v[base + (size_t)(KV_OFF + row) * 64 + col];
  }
  __syncthreads();

  const int qrow0 = blockIdx.x * 64 + wave * 16;
  if (qrow0 + 16 > Q_LEN) return;              // uniform per wave; no barriers after

  v16bf qa[2];
  {
    const u16* qp = q + base + (size_t)(Q_OFF + qrow0 + ln) * 64;
#pragma unroll
    for (int s = 0; s < 2; ++s)
      qa[s] = make_frag(qp + s * 32 + half * 8, qp + s * 32 + 16 + half * 8);
  }

  // S = q @ k^T  (16 x KV_PAD per wave)
  v8f S[KV_TILES];
#pragma unroll
  for (int j = 0; j < KV_TILES; ++j) {
    const u16* kr = &Ks[j * 16 + ln][0];
    v8f c = vzero8();
    c = wmma_bf16(qa[0], make_frag(kr + half * 8,      kr + 16 + half * 8), c);
    c = wmma_bf16(qa[1], make_frag(kr + 32 + half * 8, kr + 48 + half * 8), c);
    S[j] = c;
  }

  // softmax per q row (row lives in one VGPR across a 16-lane half)
  const float scale = 0.125f;   // d^-0.5, d=64
#pragma unroll
  for (int i = 0; i < 8; ++i) {
    float m = -3.0e38f;
#pragma unroll
    for (int j = 0; j < KV_TILES; ++j) {
      const int col = j * 16 + ln;
      float sv = (col < KV_LEN) ? S[j][i] * scale : -3.0e38f;
      S[j][i] = sv;
      m = fmaxf(m, sv);
    }
    m = fmaxf(m, __shfl_xor(m, 1));
    m = fmaxf(m, __shfl_xor(m, 2));
    m = fmaxf(m, __shfl_xor(m, 4));
    m = fmaxf(m, __shfl_xor(m, 8));
    float sum = 0.f;
#pragma unroll
    for (int j = 0; j < KV_TILES; ++j) {
      const float p = __expf(S[j][i] - m);
      S[j][i] = p;
      sum += p;
    }
    sum += __shfl_xor(sum, 1);
    sum += __shfl_xor(sum, 2);
    sum += __shfl_xor(sum, 4);
    sum += __shfl_xor(sum, 8);
    const float rinv = 1.0f / sum;
#pragma unroll
    for (int j = 0; j < KV_TILES; ++j)
      Ps[wave][half * 8 + i][j * 16 + ln] = (u16)f2bfbits(S[j][i] * rinv);
  }

  // O = P @ V  (16 x 64)
  v8f O[4];
#pragma unroll
  for (int ni = 0; ni < 4; ++ni) O[ni] = vzero8();
#pragma unroll
  for (int s = 0; s < KSTEPS; ++s) {
    const u16* pr = &Ps[wave][ln][s * 32];
    v16bf pa = make_frag(pr + half * 8, pr + 16 + half * 8);
#pragma unroll
    for (int ni = 0; ni < 4; ++ni) {
      const u16* vr = &VsT[ni * 16 + ln][s * 32];
      O[ni] = wmma_bf16(pa, make_frag(vr + half * 8, vr + 16 + half * 8), O[ni]);
    }
  }

  // write ctx [B, N, C] bf16
  const int b = bh / 12, h = bh - b * 12;
#pragma unroll
  for (int ni = 0; ni < 4; ++ni)
#pragma unroll
    for (int i = 0; i < 8; ++i) {
      const int tok = Q_OFF + qrow0 + half * 8 + i;
      const int col = h * 64 + ni * 16 + ln;
      ctx[((size_t)b * 864 + tok) * 768 + col] = (u16)f2bfbits(O[ni][i]);
    }
}

// ---------------------------------------------------------------------------
// Kernel 4: out = ctx @ W_proj + b_proj  (bf16 x bf16 -> fp32)
// A tile staged with async global->LDS copies (ASYNCcnt), B converted in-VALU.
// ---------------------------------------------------------------------------
__global__ __launch_bounds__(256) void proj_gemm_kernel(
    const u16* __restrict__ ctx, const float* __restrict__ W,
    const float* __restrict__ bias, float* __restrict__ out)
{
  __shared__ u16 As[128][40];
  __shared__ u16 Bs[128][40];

  const int tid  = threadIdx.x;
  const int lane = tid & 31;
  const int wave = tid >> 5;
  const int half = lane >> 4;
  const int ln   = lane & 15;
  const int wr   = wave >> 1;
  const int wc   = wave & 1;
  const int rowBase = blockIdx.y * 128;
  const int colBase = blockIdx.x * 128;

  const int ar = tid >> 1;
  const int ac = (tid & 1) << 4;
  const int bk = tid >> 3;
  const int bn = (tid & 7) << 4;

  v8f acc[2][4];
#pragma unroll
  for (int mi = 0; mi < 2; ++mi)
#pragma unroll
    for (int ni = 0; ni < 4; ++ni) acc[mi][ni] = vzero8();

  for (int ks = 0; ks < 24; ++ks) {
    const int k0 = ks << 5;
    { // A already bf16: async 16B copies straight into LDS
      const u16* src = ctx + (size_t)(rowBase + ar) * 768 + k0 + ac;
#if TRY_ASYNC
      async_load_b128(src,     &As[ar][ac]);
      async_load_b128(src + 8, &As[ar][ac + 8]);
#else
      const uint4* ap = (const uint4*)src;
      *(uint4*)&As[ar][ac]     = ap[0];
      *(uint4*)&As[ar][ac + 8] = ap[1];
#endif
    }
    { // W_proj fp32 -> bf16, transposed
      const float4* bp = (const float4*)(W + (size_t)(k0 + bk) * 768 + colBase + bn);
#pragma unroll
      for (int c4 = 0; c4 < 4; ++c4) {
        float4 f = bp[c4];
        Bs[bn + c4 * 4 + 0][bk] = (u16)f2bfbits(f.x);
        Bs[bn + c4 * 4 + 1][bk] = (u16)f2bfbits(f.y);
        Bs[bn + c4 * 4 + 2][bk] = (u16)f2bfbits(f.z);
        Bs[bn + c4 * 4 + 3][bk] = (u16)f2bfbits(f.w);
      }
    }
    if (ks < 23) {
      __builtin_prefetch(ctx + (size_t)(rowBase + ar) * 768 + k0 + 32 + ac, 0, 3);
      __builtin_prefetch(W + (size_t)(k0 + 32 + bk) * 768 + colBase + bn, 0, 3);
    }
#if TRY_ASYNC
    wait_async0();
#endif
    __syncthreads();

    v16bf afrag[2], bfrag[4];
#pragma unroll
    for (int mi = 0; mi < 2; ++mi) {
      const u16* r = &As[wr * 32 + mi * 16 + ln][0];
      afrag[mi] = make_frag(r + half * 8, r + 16 + half * 8);
    }
#pragma unroll
    for (int ni = 0; ni < 4; ++ni) {
      const u16* r = &Bs[wc * 64 + ni * 16 + ln][0];
      bfrag[ni] = make_frag(r + half * 8, r + 16 + half * 8);
    }
#pragma unroll
    for (int mi = 0; mi < 2; ++mi)
#pragma unroll
      for (int ni = 0; ni < 4; ++ni)
        acc[mi][ni] = wmma_bf16(afrag[mi], bfrag[ni], acc[mi][ni]);
    __syncthreads();
  }

#pragma unroll
  for (int mi = 0; mi < 2; ++mi)
#pragma unroll
    for (int ni = 0; ni < 4; ++ni) {
      const int cg = colBase + wc * 64 + ni * 16 + ln;
      const float bv = bias[cg];
#pragma unroll
      for (int i = 0; i < 8; ++i) {
        const int rg = rowBase + wr * 32 + mi * 16 + half * 8 + i;
        out[(size_t)rg * 768 + cg] = acc[mi][ni][i] + bv;
      }
    }
}

// ---------------------------------------------------------------------------
extern "C" void kernel_launch(void* const* d_in, const int* in_sizes, int n_in,
                              void* d_out, int out_size, void* d_ws, size_t ws_size,
                              hipStream_t stream) {
  (void)in_sizes; (void)n_in; (void)out_size; (void)ws_size;
  const float* x     = (const float*)d_in[0];
  const float* Wqkv  = (const float*)d_in[1];
  const float* Wproj = (const float*)d_in[2];
  const float* bproj = (const float*)d_in[3];
  float* out = (float*)d_out;

  u16* ws = (u16*)d_ws;
  const size_t E = (size_t)32 * 12 * 864 * 64;   // 21,233,664 elems each
  u16* qb  = ws;
  u16* kb  = ws + E;
  u16* vb  = ws + 2 * E;
  u16* ctx = ws + 3 * E;                          // [B, N, C] bf16 (same size)

  // qkv projection: M tiles = 27648/128 = 216, N tiles = 2304/128 = 18
  qkv_gemm_kernel<<<dim3(18, 216), dim3(256), 0, stream>>>(x, Wqkv, qb, kb, vb);

  // branch t: q rows [0,288) vs k_s/v_s rows [72,216); KV padded 144 -> 160
  attn_kernel<144, 160, 0, 288, 72>
      <<<dim3(5, 384), dim3(128), 0, stream>>>(qb, kb, vb, ctx);
  // branch s: q rows [288,864) vs k_t/v_t rows [0,72); KV padded 72 -> 96
  attn_kernel<72, 96, 288, 576, 0>
      <<<dim3(9, 384), dim3(128), 0, stream>>>(qb, kb, vb, ctx);

  // output projection: M tiles = 216, N tiles = 768/128 = 6
  proj_gemm_kernel<<<dim3(6, 216), dim3(256), 0, stream>>>(ctx, Wproj, bproj, out);
}
